// Stage_51101520888221
// MI455X (gfx1250) — matrix operand order repaired and checked by reading, not compiled, not run on hardware
//
#include <hip/hip_runtime.h>

typedef __bf16 bf16;
typedef __attribute__((ext_vector_type(16))) __bf16 v16bf;
typedef __attribute__((ext_vector_type(8)))  float  v8f;

#define N_ACT  512
#define N_OBJ  3584
#define NTOT   4096
#define DD     1028
#define DPAD   1056
#define NHID   257
#define NHPAD  272
#define NHEADS 4
#define OFS    2048
#define AFS    1024
#define NCLS   80
#define LALPHA 0.2f
#define NEG_BIG (-9e15f)
#define ATTN_ROWS 64          // i-rows per attention block (4 P tiles)
#define WCH 33                // ceil(257/8) 16-byte chunks per Wh row

union Frag { v16bf v; unsigned u[8]; };
union Chunk8 { uint4 u; bf16 e[8]; };

__device__ __forceinline__ float waveSum(float x) {
#pragma unroll
  for (int o = 16; o > 0; o >>= 1) x += __shfl_xor(x, o, 32);
  return x;
}

// ---------------- elementwise converts / inits ----------------

__global__ void convert_kernel(const float* __restrict__ in, bf16* __restrict__ out, long long n) {
  long long i = (long long)blockIdx.x * 256 + threadIdx.x;
  if (i < n) out[i] = (bf16)in[i];
}

__global__ void padrows_kernel(const float* __restrict__ X, bf16* __restrict__ Xbf) {
  long long i = (long long)blockIdx.x * 256 + threadIdx.x;
  if (i >= (long long)NTOT * DPAD) return;
  int r = (int)(i / DPAD), c = (int)(i % DPAD);
  float v = (c < DD) ? X[(size_t)r * DD + c] : 0.f;
  Xbf[i] = (bf16)v;
}

__global__ void gpad_kernel(bf16* __restrict__ Gbf) {
  int i = blockIdx.x * 256 + threadIdx.x;
  if (i >= NTOT * (DPAD - DD)) return;
  int r = i / (DPAD - DD), c = DD + i % (DPAD - DD);
  Gbf[(size_t)r * DPAD + c] = (bf16)0.f;
}

__global__ void zerof_kernel(float* p) { p[0] = 0.f; }

// ---------------- preprocessing ----------------

// mean-pool [512,1024,196] -> X[:,0:1024]; one wave per (n,c), lane-strided coalesced reads
__global__ __launch_bounds__(256) void pool_kernel(const float* __restrict__ af, float* __restrict__ X) {
  long long wid = (long long)blockIdx.x * 8 + (threadIdx.x >> 5);
  int lane = threadIdx.x & 31;
  int n = (int)(wid >> 10);
  int c = (int)(wid & 1023);
  const float* p = af + ((size_t)n * AFS + c) * 196;
  float s = 0.f;
  for (int e = lane; e < 196; e += 32) s += p[e];
  s = waveSum(s);
  if (lane == 0) X[(size_t)n * DD + c] = s * (1.f / 196.f);
}

__global__ void geo_kernel(const float* __restrict__ ab, const float* __restrict__ ob,
                           float* __restrict__ X) {
  int r = blockIdx.x * 256 + threadIdx.x;
  if (r >= NTOT) return;
  const float* b = (r < N_ACT) ? (ab + (size_t)r * 4) : (ob + (size_t)(r - N_ACT) * 4);
  float w  = b[3] - b[1];
  float h  = b[2] - b[0];
  float cx = h * 0.5f + b[0];
  float cy = w * 0.5f + b[1];
  float* o = X + (size_t)r * DD + AFS;
  o[0] = w; o[1] = h; o[2] = cx; o[3] = cy;
}

// ---------------- generic bf16 WMMA GEMM ----------------
// C[M,N] = A[M,K]bf16 * B[K,N]bf16 (+epilogue). BM=128 BN=64 BK=32, 8 waves,
// each wave owns a 32x32 C block (2x2 WMMA tiles). M must be a multiple of 128
// and A must be allocated with Kloop columns (zero padded past Kb).
template <int MODE>
__global__ __launch_bounds__(256) void gemm_kernel(
    const bf16* __restrict__ A, int ldA,
    const bf16* __restrict__ B, int ldB, long long Bsz,
    float* __restrict__ Cf, bf16* __restrict__ Cb, int ldC, long long Csz,
    const float* __restrict__ bias, const float* __restrict__ resid, int ldR,
    int Kloop, int Kb, int Nb, int Nc) {
  __shared__ bf16 sA[128 * 32];   // row-major [m][k]
  __shared__ bf16 sB[64 * 32];    // transposed [n][k]
  int z = blockIdx.z;
  const bf16* Bz = B + (long long)z * Bsz;
  int tid = threadIdx.x, lane = tid & 31, wave = tid >> 5;
  int wm = (wave >> 1) * 32, wn = (wave & 1) * 32;
  int m0 = blockIdx.y * 128, n0 = blockIdx.x * 64;
  v8f acc00 = {}, acc01 = {}, acc10 = {}, acc11 = {};
  int am = tid >> 1, ah = tid & 1;

  for (int kb = 0; kb < Kloop; kb += 32) {
    // stage A: 2x uint4 (32B) per thread, fully coalesced
    {
      const uint4* g = (const uint4*)(A + (size_t)(m0 + am) * ldA + kb + ah * 16);
      uint4 v0 = g[0];
      uint4 v1 = g[1];
      *(uint4*)&sA[am * 32 + ah * 16] = v0;
      *(uint4*)&sA[am * 32 + ah * 16 + 8] = v1;
      if (kb + 32 < Kloop)
        __builtin_prefetch(A + (size_t)(m0 + am) * ldA + kb + 32 + ah * 16, 0, 0);
    }
    // stage B transposed with bounds guards (handles ragged K/N like 1028/257)
    for (int e = tid; e < 64 * 32; e += 256) {
      int k = e >> 6, n = e & 63;
      bf16 v = (bf16)0.f;
      if (kb + k < Kb && n0 + n < Nb) v = Bz[(size_t)(kb + k) * ldB + n0 + n];
      sB[n * 32 + k] = v;
    }
    __syncthreads();

    Frag a0, a1, b0, b1;
    int mr = lane & 15, kh = lane >> 4;
#pragma unroll
    for (int j = 0; j < 8; ++j) {
      int k0 = kh * 16 + j * 2;
      a0.u[j] = *(const unsigned*)&sA[(wm + mr) * 32 + k0];
      a1.u[j] = *(const unsigned*)&sA[(wm + 16 + mr) * 32 + k0];
    }
#pragma unroll
    for (int j = 0; j < 8; ++j) {
      int k0 = kh * 16 + j * 2;
      b0.u[j] = *(const unsigned*)&sB[(wn + mr) * 32 + k0];
      b1.u[j] = *(const unsigned*)&sB[(wn + 16 + mr) * 32 + k0];
    }
    acc00 = __builtin_amdgcn_wmma_f32_16x16x32_bf16(false, a0.v, false, b0.v, (short)0, acc00, false, false);
    acc01 = __builtin_amdgcn_wmma_f32_16x16x32_bf16(false, a0.v, false, b1.v, (short)0, acc01, false, false);
    acc10 = __builtin_amdgcn_wmma_f32_16x16x32_bf16(false, a1.v, false, b0.v, (short)0, acc10, false, false);
    acc11 = __builtin_amdgcn_wmma_f32_16x16x32_bf16(false, a1.v, false, b1.v, (short)0, acc11, false, false);
    __syncthreads();
  }

  int mr = lane & 15, kh = lane >> 4;
#pragma unroll
  for (int i = 0; i < 2; ++i) {
#pragma unroll
    for (int j = 0; j < 2; ++j) {
      const v8f& acc = (i == 0) ? ((j == 0) ? acc00 : acc01)
                                : ((j == 0) ? acc10 : acc11);
#pragma unroll
      for (int r = 0; r < 8; ++r) {
        int gm = m0 + wm + i * 16 + kh * 8 + r;
        int gn = n0 + wn + j * 16 + mr;
        if (gn >= Nc) continue;
        float v = acc[r];
        if (MODE == 0) {                 // reducer: relu(C + bias)
          v = fmaxf(v + bias[gn], 0.f);
          Cf[(size_t)gm * ldC + gn] = v;
        } else if (MODE == 1) {          // bf16 store (Wh)
          (Cb + (long long)z * Csz)[(size_t)gm * ldC + gn] = (bf16)v;
        } else if (MODE == 2) {          // residual + bias
          Cf[(size_t)gm * ldC + gn] = resid[(size_t)gm * ldR + gn] + v + bias[gn];
        } else {                         // bias only (logits)
          Cf[(size_t)gm * ldC + gn] = v + bias[gn];
        }
      }
    }
  }
}

// ---------------- attention source/dest scores ----------------
// f1[h,n] = Wh[h,n,:].a1 ; f2[h,n] = Wh[h,n,:].a2 ; one wave per (h,n)
__global__ __launch_bounds__(256) void f1f2_kernel(const bf16* __restrict__ Wh,
                                                   const float* __restrict__ a,
                                                   float* __restrict__ f1,
                                                   float* __restrict__ f2) {
  int wid = blockIdx.x * 8 + (threadIdx.x >> 5);
  int lane = threadIdx.x & 31;
  int h = wid >> 12;
  int n = wid & (NTOT - 1);
  const bf16* row = Wh + (size_t)(h * NTOT + n) * NHPAD;
  const float* a1 = a + h * (2 * NHID);
  const float* a2 = a1 + NHID;
  float s1 = 0.f, s2 = 0.f;
  for (int d = lane; d < NHID; d += 32) {
    float w = (float)row[d];
    s1 += w * a1[d];
    s2 += w * a2[d];
  }
  s1 = waveSum(s1);
  s2 = waveSum(s2);
  if (lane == 0) { f1[h * NTOT + n] = s1; f2[h * NTOT + n] = s2; }
}

// online softmax stats per (h, row): masked entries use exact -9e15 semantics
__global__ __launch_bounds__(256) void rowstat_kernel(const int* __restrict__ adj,
                                                      const float* __restrict__ f1,
                                                      const float* __restrict__ f2,
                                                      float* __restrict__ rowmax,
                                                      float* __restrict__ invZ) {
  int i = blockIdx.x, tid = threadIdx.x;
  float F1[NHEADS], m[NHEADS], s[NHEADS];
#pragma unroll
  for (int h = 0; h < NHEADS; ++h) { F1[h] = f1[h * NTOT + i]; m[h] = NEG_BIG; s[h] = 0.f; }
  const int* arow = adj + (size_t)i * NTOT;
  for (int j = tid; j < NTOT; j += 256) {
    int av = arow[j];
#pragma unroll
    for (int h = 0; h < NHEADS; ++h) {
      float e = F1[h] + f2[h * NTOT + j];
      e = (e > 0.f) ? e : LALPHA * e;
      e = (av > 0) ? e : NEG_BIG;
      if (e > m[h]) { s[h] = s[h] * __expf(m[h] - e) + 1.f; m[h] = e; }
      else s[h] += __expf(e - m[h]);
    }
  }
  int lane = tid & 31, w = tid >> 5;
  __shared__ float shm[NHEADS][8], shs[NHEADS][8];
#pragma unroll
  for (int h = 0; h < NHEADS; ++h) {
    float mm = m[h], ss = s[h];
#pragma unroll
    for (int o = 16; o > 0; o >>= 1) {
      float om = __shfl_xor(mm, o, 32);
      float os = __shfl_xor(ss, o, 32);
      float M = fmaxf(mm, om);
      ss = ss * __expf(mm - M) + os * __expf(om - M);
      mm = M;
    }
    if (lane == 0) { shm[h][w] = mm; shs[h][w] = ss; }
  }
  __syncthreads();
  if (tid < NHEADS) {
    int h = tid;
    float mm = shm[h][0], ss = shs[h][0];
    for (int q = 1; q < 8; ++q) {
      float om = shm[h][q], os = shs[h][q];
      float M = fmaxf(mm, om);
      ss = ss * __expf(mm - M) + os * __expf(om - M);
      mm = M;
    }
    rowmax[h * NTOT + i] = mm;
    invZ[h * NTOT + i] = 1.f / ss;
  }
}

// fused attention: 64 i-rows per block (4 P tiles) so each staged Wh tile is
// reused 4x. Staging reads uint4 (8 bf16) per global op and scatters b16 into
// the transposed LDS tile. Each wave owns two compile-time d-tiles (d 0..255);
// column d=256 runs on a scalar FMA path. All register arrays have
// compile-time trip counts -> no dynamic VGPR indexing, no spills.
__global__ __launch_bounds__(256) void attn_kernel(
    const int* __restrict__ adj, const float* __restrict__ f1,
    const float* __restrict__ f2, const float* __restrict__ rowmax,
    const float* __restrict__ invZ, const bf16* __restrict__ Wh,
    bf16* __restrict__ Gbf) {
  __shared__ bf16 sP[ATTN_ROWS * 32];  // P tiles, row-major [m][k]
  __shared__ bf16 sW[NHID * 32];       // Wh tile, transposed [d][k], d<257
  int i0 = blockIdx.x * ATTN_ROWS;
  int h = blockIdx.y;
  int tid = threadIdx.x, lane = tid & 31, wave = tid >> 5;
  int pm = tid >> 4;                   // base P row; thread covers pm + q*16
  int pj = (tid & 15) * 2;             // P col pair within j-tile
  int mr = lane & 15, kh = lane >> 4;
  int t0 = wave * 2, t1 = t0 + 1;      // this wave's d-tiles
  float F1[4], RM[4], IZ[4];
#pragma unroll
  for (int q = 0; q < 4; ++q) {
    int row = i0 + pm + q * 16;
    F1[q] = f1[h * NTOT + row];
    RM[q] = rowmax[h * NTOT + row];
    IZ[q] = invZ[h * NTOT + row];
  }
  v8f acc[4][2] = {};
  float accL[4] = {};                  // column d=256 accumulators

  for (int jb = 0; jb < NTOT; jb += 32) {
    // stage Wh tile transposed: 16B global chunks, b16 LDS scatter
    for (int c = tid; c < WCH * 32; c += 256) {
      int k = c / WCH, ch = c - k * WCH;
      Chunk8 vv;
      vv.u = *(const uint4*)(Wh + ((size_t)(h * NTOT + jb + k)) * NHPAD + ch * 8);
      int d0 = ch * 8;
#pragma unroll
      for (int q = 0; q < 8; ++q)
        if (d0 + q < NHID) sW[(d0 + q) * 32 + k] = vv.e[q];
    }
    if (jb + 32 < NTOT)
      __builtin_prefetch(Wh + ((size_t)(h * NTOT + jb + 32 + (tid >> 3))) * NHPAD, 0, 0);

    // compute P tiles (exact -9e15 masking semantics)
    float f2j0 = f2[h * NTOT + jb + pj];
    float f2j1 = f2[h * NTOT + jb + pj + 1];
    float pv[4][2];
#pragma unroll
    for (int q = 0; q < 4; ++q) {
      int row = i0 + pm + q * 16;
      int av0 = adj[(size_t)row * NTOT + jb + pj];
      int av1 = adj[(size_t)row * NTOT + jb + pj + 1];
      float e0 = F1[q] + f2j0;
      float e1 = F1[q] + f2j1;
      e0 = (e0 > 0.f) ? e0 : LALPHA * e0;
      e1 = (e1 > 0.f) ? e1 : LALPHA * e1;
      e0 = (av0 > 0) ? e0 : NEG_BIG;
      e1 = (av1 > 0) ? e1 : NEG_BIG;
      pv[q][0] = __expf(e0 - RM[q]) * IZ[q];
      pv[q][1] = __expf(e1 - RM[q]) * IZ[q];
      sP[(pm + q * 16) * 32 + pj] = (bf16)pv[q][0];
      sP[(pm + q * 16) * 32 + pj + 1] = (bf16)pv[q][1];
    }
    __syncthreads();

    // scalar tail: column d=256 (uses staged sW)
    {
      float w0 = (float)sW[256 * 32 + pj];
      float w1 = (float)sW[256 * 32 + pj + 1];
#pragma unroll
      for (int q = 0; q < 4; ++q) accL[q] += pv[q][0] * w0 + pv[q][1] * w1;
    }

    Frag a[4], b0, b1;
#pragma unroll
    for (int q = 0; q < 4; ++q)
#pragma unroll
      for (int j = 0; j < 8; ++j) {
        int k0 = kh * 16 + j * 2;
        a[q].u[j] = *(const unsigned*)&sP[(q * 16 + mr) * 32 + k0];
      }
#pragma unroll
    for (int j = 0; j < 8; ++j) {
      int k0 = kh * 16 + j * 2;
      b0.u[j] = *(const unsigned*)&sW[(t0 * 16 + mr) * 32 + k0];
      b1.u[j] = *(const unsigned*)&sW[(t1 * 16 + mr) * 32 + k0];
    }
#pragma unroll
    for (int q = 0; q < 4; ++q) {
      acc[q][0] = __builtin_amdgcn_wmma_f32_16x16x32_bf16(false, a[q].v, false, b0.v, (short)0, acc[q][0], false, false);
      acc[q][1] = __builtin_amdgcn_wmma_f32_16x16x32_bf16(false, a[q].v, false, b1.v, (short)0, acc[q][1], false, false);
    }
    __syncthreads();
  }

  // epilogue: elu + store WMMA tiles (d always < 256 here)
#pragma unroll
  for (int q = 0; q < 4; ++q) {
#pragma unroll
    for (int r = 0; r < 8; ++r) {
      int m = i0 + q * 16 + kh * 8 + r;
      float v0 = acc[q][0][r];
      v0 = (v0 > 0.f) ? v0 : (__expf(v0) - 1.f);
      Gbf[(size_t)m * DPAD + h * NHID + t0 * 16 + mr] = (bf16)v0;
      float v1 = acc[q][1][r];
      v1 = (v1 > 0.f) ? v1 : (__expf(v1) - 1.f);
      Gbf[(size_t)m * DPAD + h * NHID + t1 * 16 + mr] = (bf16)v1;
    }
  }
  // column d=256: reduce over the 16 threads sharing each P row (half-wave)
#pragma unroll
  for (int q = 0; q < 4; ++q) {
    float aL = accL[q];
#pragma unroll
    for (int o = 8; o > 0; o >>= 1) aL += __shfl_xor(aL, o, 16);
    if ((tid & 15) == 0) {
      float v = (aL > 0.f) ? aL : (__expf(aL) - 1.f);
      Gbf[(size_t)(i0 + pm + q * 16) * DPAD + h * NHID + 256] = (bf16)v;
    }
  }
}

// ---------------- layernorm (+ bf16 repack with zero pad) ----------------

__global__ __launch_bounds__(256) void ln_kernel(const float* __restrict__ X2,
                                                 const float* __restrict__ g,
                                                 const float* __restrict__ b,
                                                 float* __restrict__ X,
                                                 bf16* __restrict__ Xbf) {
  int r = blockIdx.x, tid = threadIdx.x;
  const float* row = X2 + (size_t)r * DD;
  float s1 = 0.f, s2 = 0.f;
  for (int c = tid; c < DD; c += 256) { float v = row[c]; s1 += v; s2 += v * v; }
  s1 = waveSum(s1);
  s2 = waveSum(s2);
  __shared__ float sh1[8], sh2[8];
  int lane = tid & 31, w = tid >> 5;
  if (lane == 0) { sh1[w] = s1; sh2[w] = s2; }
  __syncthreads();
  float t1 = 0.f, t2 = 0.f;
#pragma unroll
  for (int q = 0; q < 8; ++q) { t1 += sh1[q]; t2 += sh2[q]; }
  float mean = t1 * (1.f / DD);
  float var = t2 * (1.f / DD) - mean * mean;
  float rs = rsqrtf(var + 1e-5f);
  for (int c = tid; c < DPAD; c += 256) {
    if (c < DD) {
      float y = (row[c] - mean) * rs * g[c] + b[c];
      X[(size_t)r * DD + c] = y;
      Xbf[(size_t)r * DPAD + c] = (bf16)y;
    } else {
      Xbf[(size_t)r * DPAD + c] = (bf16)0.f;
    }
  }
}

// ---------------- sigmoid + BCE loss ----------------

__global__ __launch_bounds__(256) void head_kernel(const float* __restrict__ logits,
                                                   const float* __restrict__ labels,
                                                   float* __restrict__ out,
                                                   float* __restrict__ loss) {
  int i = blockIdx.x * 256 + threadIdx.x;   // exactly 512*80 threads
  float lg = logits[i], y = labels[i];
  out[i] = 1.f / (1.f + __expf(-lg));
  float t = fmaxf(lg, 0.f) - lg * y + log1pf(__expf(-fabsf(lg)));
  t = waveSum(t);
  __shared__ float sh[8];
  int lane = threadIdx.x & 31, w = threadIdx.x >> 5;
  if (lane == 0) sh[w] = t;
  __syncthreads();
  if (threadIdx.x == 0) {
    float s = 0.f;
    for (int q = 0; q < 8; ++q) s += sh[q];
    atomicAdd(loss, s * (1.f / (N_ACT * NCLS)));
  }
}

// ---------------- host orchestration ----------------

extern "C" void kernel_launch(void* const* d_in, const int* in_sizes, int n_in,
                              void* d_out, int out_size, void* d_ws, size_t ws_size,
                              hipStream_t stream) {
  (void)in_sizes; (void)n_in; (void)out_size; (void)ws_size;
  const float* aF   = (const float*)d_in[0];
  const float* aL   = (const float*)d_in[1];
  const float* aB   = (const float*)d_in[2];
  const float* oF   = (const float*)d_in[3];
  const float* oB   = (const float*)d_in[4];
  const int*   adj  = (const int*)d_in[5];
  const float* objW = (const float*)d_in[6];
  const float* objb = (const float*)d_in[7];
  const float* g1W  = (const float*)d_in[8];
  const float* g1a  = (const float*)d_in[9];
  const float* fc1W = (const float*)d_in[10];
  const float* fc1b = (const float*)d_in[11];
  const float* ln1g = (const float*)d_in[12];
  const float* ln1b = (const float*)d_in[13];
  const float* g2W  = (const float*)d_in[14];
  const float* g2a  = (const float*)d_in[15];
  const float* fc2W = (const float*)d_in[16];
  const float* fc2b = (const float*)d_in[17];
  const float* ln2g = (const float*)d_in[18];
  const float* ln2b = (const float*)d_in[19];
  const float* logW = (const float*)d_in[20];
  const float* logb = (const float*)d_in[21];
  float* out = (float*)d_out;

  size_t off = 0;
  auto alloc = [&](size_t bytes) -> void* {
    void* p = (char*)d_ws + off;
    off = (off + bytes + 255) & ~(size_t)255;
    return p;
  };
  float* X      = (float*)alloc(sizeof(float) * (size_t)NTOT * DD);
  float* X2     = (float*)alloc(sizeof(float) * (size_t)NTOT * DD);
  bf16* Xbf     = (bf16*)alloc(2ull * NTOT * DPAD);
  bf16* Gbf     = (bf16*)alloc(2ull * NTOT * DPAD);
  bf16* OFbf    = (bf16*)alloc(2ull * N_OBJ * OFS);
  bf16* objWbf  = (bf16*)alloc(2ull * OFS * AFS);
  bf16* g1Wbf   = (bf16*)alloc(2ull * NHEADS * DD * NHID);
  bf16* g2Wbf   = (bf16*)alloc(2ull * NHEADS * DD * NHID);
  bf16* fc1Wbf  = (bf16*)alloc(2ull * DD * DD);
  bf16* fc2Wbf  = (bf16*)alloc(2ull * DD * DD);
  bf16* logWbf  = (bf16*)alloc(2ull * DD * NCLS);
  bf16* Whbf    = (bf16*)alloc(2ull * NHEADS * NTOT * NHPAD);
  float* f1     = (float*)alloc(sizeof(float) * NHEADS * NTOT);
  float* f2     = (float*)alloc(sizeof(float) * NHEADS * NTOT);
  float* rmax   = (float*)alloc(sizeof(float) * NHEADS * NTOT);
  float* rinv   = (float*)alloc(sizeof(float) * NHEADS * NTOT);
  float* logits = (float*)alloc(sizeof(float) * N_ACT * NCLS);

  auto cvt = [&](const float* src, bf16* dst, long long n) {
    convert_kernel<<<(unsigned)((n + 255) / 256), 256, 0, stream>>>(src, dst, n);
  };
  cvt(oF, OFbf, (long long)N_OBJ * OFS);
  cvt(objW, objWbf, (long long)OFS * AFS);
  cvt(g1W, g1Wbf, (long long)NHEADS * DD * NHID);
  cvt(g2W, g2Wbf, (long long)NHEADS * DD * NHID);
  cvt(fc1W, fc1Wbf, (long long)DD * DD);
  cvt(fc2W, fc2Wbf, (long long)DD * DD);
  cvt(logW, logWbf, (long long)DD * NCLS);

  pool_kernel<<<65536, 256, 0, stream>>>(aF, X);
  geo_kernel<<<(NTOT + 255) / 256, 256, 0, stream>>>(aB, oB, X);
  // object reducer: relu(OF @ objW + b) -> X rows 512..4095
  gemm_kernel<0><<<dim3(AFS / 64, N_OBJ / 128, 1), 256, 0, stream>>>(
      OFbf, OFS, objWbf, AFS, 0,
      X + (size_t)N_ACT * DD, nullptr, DD, 0,
      objb, nullptr, 0, OFS, OFS, AFS, AFS);
  padrows_kernel<<<(unsigned)(((long long)NTOT * DPAD + 255) / 256), 256, 0, stream>>>(X, Xbf);
  gpad_kernel<<<(NTOT * (DPAD - DD) + 255) / 256, 256, 0, stream>>>(Gbf);

  for (int layer = 0; layer < 2; ++layer) {
    const bf16* gW  = layer ? g2Wbf : g1Wbf;
    const float* ga = layer ? g2a : g1a;
    const bf16* fW  = layer ? fc2Wbf : fc1Wbf;
    const float* fb = layer ? fc2b : fc1b;
    const float* lg = layer ? ln2g : ln1g;
    const float* lb = layer ? ln2b : ln1b;
    // Wh[h] = X @ W[h]  (batched over heads via grid.z)
    gemm_kernel<1><<<dim3((NHPAD + 63) / 64, NTOT / 128, NHEADS), 256, 0, stream>>>(
        Xbf, DPAD, gW, NHID, (long long)DD * NHID,
        nullptr, Whbf, NHPAD, (long long)NTOT * NHPAD,
        nullptr, nullptr, 0, DPAD, DD, NHID, NHPAD);
    f1f2_kernel<<<(NHEADS * NTOT) / 8, 256, 0, stream>>>(Whbf, ga, f1, f2);
    rowstat_kernel<<<NTOT, 256, 0, stream>>>(adj, f1, f2, rmax, rinv);
    attn_kernel<<<dim3(NTOT / ATTN_ROWS, NHEADS), 256, 0, stream>>>(adj, f1, f2, rmax, rinv, Whbf, Gbf);
    // X2 = X + G @ fcW + fcb
    gemm_kernel<2><<<dim3((DD + 63) / 64, NTOT / 128, 1), 256, 0, stream>>>(
        Gbf, DPAD, fW, DD, 0,
        X2, nullptr, DD, 0,
        fb, X, DD, DPAD, DD, DD, DD);
    ln_kernel<<<NTOT, 256, 0, stream>>>(X2, lg, lb, X, Xbf);
  }

  gemm_kernel<3><<<dim3((NCLS + 63) / 64, N_ACT / 128, 1), 256, 0, stream>>>(
      Xbf, DPAD, logWbf, NCLS, 0,
      logits, nullptr, NCLS, 0,
      logb, nullptr, 0, DPAD, DD, NCLS, NCLS);

  zerof_kernel<<<1, 1, 0, stream>>>(out + N_ACT * NCLS);
  head_kernel<<<(N_ACT * NCLS) / 256, 256, 0, stream>>>(logits, aL, out, out + N_ACT * NCLS);
}
